// HetGTCN_LW_76682346102824
// MI455X (gfx1250) — matrix operand order repaired and checked by reading, not compiled, not run on hardware
//
#include <hip/hip_runtime.h>
#include <hip/hip_bf16.h>
#include <math.h>

typedef __attribute__((ext_vector_type(16))) _Float16 v16h;
typedef __attribute__((ext_vector_type(8)))  _Float16 v8h;
typedef __attribute__((ext_vector_type(8)))  float    v8f;

#define DIN  256
#define DH   128
#define DOUT 8
#define HOP  5

// ---------------------------------------------------------------------------
// Softmax of lw pairs: lw[5,4] -> wsoft[5,4] where (0,1) and (2,3) are pairs.
// ---------------------------------------------------------------------------
__global__ void softmax_lw_kernel(const float* __restrict__ lw,
                                  float* __restrict__ wsoft) {
  int t = threadIdx.x;                 // 0..9 pairs
  if (t < HOP * 2) {
    int base = (t >> 1) * 4 + (t & 1) * 2;
    float a = lw[base], b = lw[base + 1];
    float m = fmaxf(a, b);
    float ea = __expf(a - m), eb = __expf(b - m);
    float inv = 1.0f / (ea + eb);
    wsoft[base]     = ea * inv;
    wsoft[base + 1] = eb * inv;
  }
}

// ---------------------------------------------------------------------------
// Convert W1 [DIN, DH] f32 (row-major, k-major rows) to f16 transposed
// WT [DH, DIN] so each WMMA B-fragment lane reads two contiguous 16B runs.
// ---------------------------------------------------------------------------
__global__ void convert_w1_kernel(const float* __restrict__ W,
                                  _Float16* __restrict__ WT) {
  int tid = blockIdx.x * blockDim.x + threadIdx.x;   // DIN*DH threads
  if (tid < DIN * DH) {
    int n = tid % DH;
    int k = tid / DH;
    WT[(size_t)n * DIN + k] = (_Float16)W[(size_t)k * DH + n];
  }
}

// ---------------------------------------------------------------------------
// Input projection: out = relu(X @ W1 + b1), written to two buffers
// (x_proj kept for diag terms, h0 as the mutable hop state).
// Block: 256 threads = 8 waves; block computes 16 rows x 128 cols.
// Wave j computes the 16x16 tile at columns [16j,16j+16), K=256 in 8 WMMA
// steps of 32 using v_wmma_f32_16x16x32_f16 (fp32 accumulate).
// ---------------------------------------------------------------------------
__global__ __launch_bounds__(256)
void proj_wmma_kernel(const float* __restrict__ X,
                      const _Float16* __restrict__ WT,   // [DH, DIN] f16
                      const float* __restrict__ bias,
                      float* __restrict__ out0,
                      float* __restrict__ out1) {
  __shared__ __align__(64) _Float16 As[16 * DIN];     // 8 KB, f16 A-slab

  const int row0 = blockIdx.x * 16;

  // Stage + convert 16x256 A slab (f32 -> f16), fully coalesced.
  for (int t = threadIdx.x; t < 16 * DIN; t += 256) {
    int m = t >> 8;          // DIN == 256
    int k = t & (DIN - 1);
    As[t] = (_Float16)X[(size_t)(row0 + m) * DIN + k];
  }
  __syncthreads();

  const int lane = threadIdx.x & 31;
  const int j    = threadIdx.x >> 5;          // wave id = column tile
  const int mlo  = lane & 15;                 // A row within tile
  const int n    = j * 16 + (lane & 15);      // B/D column
  const int hi   = lane >> 4;                 // 0: lanes 0-15, 1: lanes 16-31
  const int qa   = hi * 8;                    // A K-chunk base within 32
  const int qb   = hi * 8;                    // B K-chunk base within 32

  v8f acc = {};
#pragma unroll
  for (int k0 = 0; k0 < DIN; k0 += 32) {
    // A 16x32 f16 fragment: lane holds K=[qa..qa+7] and K=[qa+16..qa+23]
    const v8h alo = *(const v8h*)&As[mlo * DIN + k0 + qa];
    const v8h ahi = *(const v8h*)&As[mlo * DIN + k0 + qa + 16];
    v16h a;
#pragma unroll
    for (int t = 0; t < 8; ++t) { a[t] = alo[t]; a[t + 8] = ahi[t]; }
    // B 32x16 f16 fragment: lane = column n, K=[qb..qb+7] and K=[qb+16..qb+23]
    const v8h blo = *(const v8h*)&WT[(size_t)n * DIN + k0 + qb];
    const v8h bhi = *(const v8h*)&WT[(size_t)n * DIN + k0 + qb + 16];
    v16h b;
#pragma unroll
    for (int t = 0; t < 8; ++t) { b[t] = blo[t]; b[t + 8] = bhi[t]; }
    acc = __builtin_amdgcn_wmma_f32_16x16x32_f16(
        /*neg_a=*/false, a, /*neg_b=*/false, b,
        /*c_mod=*/(short)0, acc, /*reuse_a=*/false, /*reuse_b=*/false);
  }

  const float bn = bias[n];
#pragma unroll
  for (int r = 0; r < 8; ++r) {
    int m = r + hi * 8;                        // D layout: VGPR r -> M=r / M=8+r
    float v = acc[r] + bn;
    v = v > 0.0f ? v : 0.0f;                   // ReLU
    size_t o = (size_t)(row0 + m) * DH + n;
    out0[o] = v;
    out1[o] = v;
  }
}

// ---------------------------------------------------------------------------
// h_new = (w0*diagA + w1*diagB) * x_proj   (folds both diag terms; acts as
// the initializer so the SpMMs can accumulate without a zeroing pass)
// ---------------------------------------------------------------------------
__global__ __launch_bounds__(256)
void init_diag_kernel(float* __restrict__ hnew,
                      const float* __restrict__ xproj,
                      const float* __restrict__ dA,
                      const float* __restrict__ dB,
                      const float* __restrict__ wpair,
                      int total) {
  int i = blockIdx.x * blockDim.x + threadIdx.x;
  if (i < total) {
    int node = i >> 7;                        // DH == 128
    float s = wpair[0] * dA[node] + wpair[1] * dB[node];
    hnew[i] = s * xproj[i];
  }
}

// ---------------------------------------------------------------------------
// Scaled SpMM with scatter-add: hout[rows[e]] += w*vals[e]*hin[cols[e]].
// One wave per edge. The edge index is forced wave-uniform via
// readfirstlane so edge metadata comes in through scalar loads (KMcnt) and
// the gather row base is SGPR-held (SADDR-form VMEM with only
// lane*16B as the vector offset). Each lane gathers one float4
// (global_load_b128) of the 512B source row and issues 4 no-return
// global_atomic_add_f32 at immediate offsets, scope DEV. h buffers
// (51.2 MB) are L2-resident on MI455X (192 MB L2), so both the gathers and
// the atomic adds are serviced by L2's atomic units.
// ---------------------------------------------------------------------------
__global__ __launch_bounds__(256)
void spmm_atomic_kernel(const int* __restrict__ rows,
                        const int* __restrict__ cols,
                        const float* __restrict__ vals,
                        const float* __restrict__ hin,
                        float* __restrict__ hout,
                        const float* __restrict__ wptr,
                        int nedges) {
  const float w = *wptr;
  const int lane = threadIdx.x & 31;
  // Force wave-uniform edge index into SGPRs.
  const int waveInBlk = __builtin_amdgcn_readfirstlane(threadIdx.x >> 5);
  const int wavesPerBlk = blockDim.x >> 5;
  const int wave   = blockIdx.x * wavesPerBlk + waveInBlk;
  const int nwaves = gridDim.x * wavesPerBlk;

  for (int e = wave; e < nedges; e += nwaves) {
    const int r = rows[e];                    // uniform -> s_load
    const int c = cols[e];                    // uniform -> s_load
    const float s = w * vals[e];              // uniform -> s_load
    const float4 v = ((const float4*)(hin + (size_t)c * DH))[lane];
    float* dst = hout + (size_t)r * DH + lane * 4;
    float d0 = s * v.x, d1 = s * v.y, d2 = s * v.z, d3 = s * v.w;
    // 4 no-return fp32 atomics at immediate offsets from one address pair.
    asm volatile(
        "global_atomic_add_f32 %0, %1, off scope:SCOPE_DEV\n\t"
        "global_atomic_add_f32 %0, %2, off offset:4 scope:SCOPE_DEV\n\t"
        "global_atomic_add_f32 %0, %3, off offset:8 scope:SCOPE_DEV\n\t"
        "global_atomic_add_f32 %0, %4, off offset:12 scope:SCOPE_DEV"
        :
        : "v"(dst), "v"(d0), "v"(d1), "v"(d2), "v"(d3)
        : "memory");
  }
}

// ---------------------------------------------------------------------------
// Output head: out = h_p @ W2 + b2, W2 [128,8] staged in LDS.
// Thread per output element (row, col).
// ---------------------------------------------------------------------------
__global__ __launch_bounds__(256)
void out_gemm_kernel(const float* __restrict__ h,
                     const float* __restrict__ W2,
                     const float* __restrict__ b2,
                     float* __restrict__ out,
                     int total) {
  __shared__ float w2s[DH * DOUT];            // 4 KB
  for (int t = threadIdx.x; t < DH * DOUT; t += 256) w2s[t] = W2[t];
  __syncthreads();

  int i = blockIdx.x * blockDim.x + threadIdx.x;
  if (i < total) {
    int row = i >> 3;
    int col = i & 7;
    float acc = b2[col];
    const float* hr = h + (size_t)row * DH;
#pragma unroll 4
    for (int k = 0; k < DH; ++k) acc = fmaf(hr[k], w2s[k * DOUT + col], acc);
    out[i] = acc;
  }
}

// ---------------------------------------------------------------------------
// Orchestration
// ---------------------------------------------------------------------------
extern "C" void kernel_launch(void* const* d_in, const int* in_sizes, int n_in,
                              void* d_out, int out_size, void* d_ws, size_t ws_size,
                              hipStream_t stream) {
  const int N = in_sizes[0] / DIN;            // 100000
  const int E = in_sizes[2];                  // 1600000

  const float* x_paper  = (const float*)d_in[0];
  const float* x_author = (const float*)d_in[1];
  // edge type blocks: pp=2.., pa=6.., ap=10.., aa=14..
  const int*   rows_pp = (const int*)d_in[2];
  const int*   cols_pp = (const int*)d_in[3];
  const float* vals_pp = (const float*)d_in[4];
  const float* diag_pp = (const float*)d_in[5];
  const int*   rows_pa = (const int*)d_in[6];
  const int*   cols_pa = (const int*)d_in[7];
  const float* vals_pa = (const float*)d_in[8];
  const float* diag_pa = (const float*)d_in[9];
  const int*   rows_ap = (const int*)d_in[10];
  const int*   cols_ap = (const int*)d_in[11];
  const float* vals_ap = (const float*)d_in[12];
  const float* diag_ap = (const float*)d_in[13];
  const int*   rows_aa = (const int*)d_in[14];
  const int*   cols_aa = (const int*)d_in[15];
  const float* vals_aa = (const float*)d_in[16];
  const float* diag_aa = (const float*)d_in[17];
  const float* W1_paper  = (const float*)d_in[18];
  const float* b1_paper  = (const float*)d_in[19];
  const float* W1_author = (const float*)d_in[20];
  const float* b1_author = (const float*)d_in[21];
  const float* W2 = (const float*)d_in[22];
  const float* b2 = (const float*)d_in[23];
  const float* lw = (const float*)d_in[24];

  // ---- scratch layout (floats) ----
  const size_t NDH = (size_t)N * DH;
  float* ws      = (float*)d_ws;
  float* xp_proj = ws;                 // N*DH
  float* xa_proj = xp_proj + NDH;      // N*DH
  float* hp0     = xa_proj + NDH;      // N*DH  (ping-pong paper state)
  float* hp1     = hp0 + NDH;
  float* ha0     = hp1 + NDH;          // ping-pong author state
  float* ha1     = ha0 + NDH;
  float* wsoft   = ha1 + NDH;          // HOP*4 (+pad)
  _Float16* W1pT = (_Float16*)(wsoft + 32);          // DH*DIN halves
  _Float16* W1aT = W1pT + (size_t)DH * DIN;

  // 1. softmax of layer weights
  softmax_lw_kernel<<<1, 32, 0, stream>>>(lw, wsoft);

  // 2. weight convert/transpose to f16
  {
    int nt = DIN * DH;
    int blks = (nt + 255) / 256;
    convert_w1_kernel<<<blks, 256, 0, stream>>>(W1_paper,  W1pT);
    convert_w1_kernel<<<blks, 256, 0, stream>>>(W1_author, W1aT);
  }

  // 3. WMMA input projections (write x_proj and h0)
  {
    int blks = N / 16;                 // 100000 / 16 = 6250
    proj_wmma_kernel<<<blks, 256, 0, stream>>>(x_paper,  W1pT, b1_paper,
                                               xp_proj, hp0);
    proj_wmma_kernel<<<blks, 256, 0, stream>>>(x_author, W1aT, b1_author,
                                               xa_proj, ha0);
  }

  const int totNDH   = (int)NDH;
  const int initBlks = (totNDH + 255) / 256;
  const int spmmBlks = 4096;           // grid-stride, 32768 waves

  float* hp_old = hp0; float* hp_new = hp1;
  float* ha_old = ha0; float* ha_new = ha1;

  // 4. HOP iterations (stream order provides the cross-kernel dependencies)
  for (int hop = 0; hop < HOP; ++hop) {
    const float* wp = wsoft + hop * 4;       // [w_p0, w_p1, w_a0, w_a1]

    // paper update: h_p_new = w0*spmm_pp(h_p) + w1*spmm_pa(h_a)
    //               + (w0*diag_pp + w1*diag_pa)*x_p
    init_diag_kernel<<<initBlks, 256, 0, stream>>>(hp_new, xp_proj,
                                                   diag_pp, diag_pa,
                                                   wp, totNDH);
    spmm_atomic_kernel<<<spmmBlks, 256, 0, stream>>>(rows_pp, cols_pp, vals_pp,
                                                     hp_old, hp_new, wp + 0, E);
    spmm_atomic_kernel<<<spmmBlks, 256, 0, stream>>>(rows_pa, cols_pa, vals_pa,
                                                     ha_old, hp_new, wp + 1, E);

    // author update (uses the freshly updated h_p_new)
    init_diag_kernel<<<initBlks, 256, 0, stream>>>(ha_new, xa_proj,
                                                   diag_ap, diag_aa,
                                                   wp + 2, totNDH);
    spmm_atomic_kernel<<<spmmBlks, 256, 0, stream>>>(rows_ap, cols_ap, vals_ap,
                                                     hp_new, ha_new, wp + 2, E);
    spmm_atomic_kernel<<<spmmBlks, 256, 0, stream>>>(rows_aa, cols_aa, vals_aa,
                                                     ha_old, ha_new, wp + 3, E);

    // ping-pong
    float* t;
    t = hp_old; hp_old = hp_new; hp_new = t;
    t = ha_old; ha_old = ha_new; ha_new = t;
  }

  // 5. output head
  {
    int tot = N * DOUT;
    int blks = (tot + 255) / 256;
    out_gemm_kernel<<<blks, 256, 0, stream>>>(hp_old, W2, b2,
                                              (float*)d_out, tot);
  }
}
